// Model_49022756716979
// MI455X (gfx1250) — matrix operand order repaired and checked by reading, not compiled, not run on hardware
//
#include <hip/hip_runtime.h>
#include <math.h>

#define N1C 8
#define N2C 8
#define PC  1024
#define DC  768
#define PHC 32
#define HC  512

#define BM 128
#define BN 128
#define BK 32
#define NK (DC / BK)   // 24 k-steps
#define LDA 40   // padded LDS row stride in halves (64B data + 16B pad per row)

typedef __attribute__((ext_vector_type(16))) _Float16 v16h;
typedef __attribute__((ext_vector_type(8)))  _Float16 v8h;
typedef __attribute__((ext_vector_type(8)))  float    v8f;
typedef __attribute__((ext_vector_type(4)))  unsigned int v4u;
typedef __attribute__((ext_vector_type(8)))  unsigned int v8u;

// ---------------------------------------------------------------------------
// Kernel 1: L2-normalize each D=768 row; emit f16 copy + fp32 squared norm.
// ---------------------------------------------------------------------------
__global__ void normalize_rows(const float* __restrict__ in,
                               _Float16* __restrict__ outh,
                               float* __restrict__ a2) {
  const int row = blockIdx.x;
  const int tid = threadIdx.x;
  const float* src = in + (size_t)row * DC;
  float x0 = src[tid];
  float x1 = src[tid + 256];
  float x2 = src[tid + 512];
  float ss = x0 * x0 + x1 * x1 + x2 * x2;
  #pragma unroll
  for (int off = 16; off > 0; off >>= 1) ss += __shfl_xor(ss, off, 32);
  __shared__ float red[8];
  const int lane = tid & 31, wid = tid >> 5;
  if (lane == 0) red[wid] = ss;
  __syncthreads();
  float tot = 0.f;
  #pragma unroll
  for (int i = 0; i < 8; ++i) tot += red[i];
  const float rn = rsqrtf(tot);
  _Float16* dst = outh + (size_t)row * DC;
  dst[tid]       = (_Float16)(x0 * rn);
  dst[tid + 256] = (_Float16)(x1 * rn);
  dst[tid + 512] = (_Float16)(x2 * rn);
  if (tid == 0) a2[row] = tot * rn * rn;
}

// ---------------------------------------------------------------------------
// TDM: DMA one 32(half)x128(row) tile, row stride DC halves, global -> LDS,
// writing the padded LDS layout (16 DWORDs data + 4 DWORDs pad per row
// == LDA=40 halves). 2D descriptor: groups 0 and 1 only (VADDR2/3 = NULL).
// D# layout per CDNA5 ISA.
// ---------------------------------------------------------------------------
__device__ __forceinline__ void tdm_load_tile32x128(const _Float16* gsrc,
                                                    unsigned int lds_off) {
  unsigned long long ga = (unsigned long long)(uintptr_t)gsrc;
  v4u g0;
  g0[0] = 1u;                                    // count=1, user descriptor
  g0[1] = lds_off;                               // lds_addr (bytes)
  g0[2] = (unsigned int)ga;                      // global_addr[31:0]
  g0[3] = (unsigned int)((ga >> 32) & 0x01FFFFFFu) | (2u << 30); // [56:32] + type=2
  v8u g1;
  g1[0] = (1u << 16)        // data_size = 1 -> 2-byte elements
        | (1u << 20)        // pad_enable
        | (3u << 22)        // pad_interval: 16 DWORDs (one 64B row) before pad
        | (3u << 25);       // pad_amount: 4 DWORDs (16B) -> LDS stride 80B
  g1[1] = ((unsigned int)DC & 0xFFFFu) << 16;                    // tensor_dim0 lo16
  g1[2] = ((unsigned int)DC >> 16) | (((unsigned int)PC & 0xFFFFu) << 16); // dim0 hi16, dim1 lo16
  g1[3] = ((unsigned int)PC >> 16) | (32u << 16);                // dim1 hi16, tile_dim0=32
  g1[4] = 128u;                                                  // tile_dim1=128, tile_dim2=0
  g1[5] = (unsigned int)DC;                                      // tensor_dim0_stride lo32
  g1[6] = 0u;                                                    // stride hi16, dim1_stride lo16
  g1[7] = 0u;
  asm volatile("tensor_load_to_lds %0, %1" :: "s"(g0), "s"(g1) : "memory");
}

// ---------------------------------------------------------------------------
// Kernel 2: fused GEMM + min-over-q, TDM-staged and double-buffered.
// Block = 256 threads (8 wave32). Wave 0 drives the Tensor Data Mover to
// prefetch tile k+1 into the alternate LDS buffer while all waves run WMMA
// on tile k. One barrier per k-step.
// ---------------------------------------------------------------------------
__global__ void __launch_bounds__(256)
gemm_min_kernel(const _Float16* __restrict__ fh,
                const _Float16* __restrict__ gh,
                const float* __restrict__ fa2,
                const float* __restrict__ gb2,
                float* __restrict__ patch) {
  __shared__ _Float16 As[2][BM * LDA];
  __shared__ _Float16 Bs[2][BN * LDA];

  const int pt = blockIdx.x;        // p-tile: 0..7
  const int nm = blockIdx.y;        // 0..63
  const int n1 = nm >> 3;
  const int n2 = nm & 7;

  const int tid  = threadIdx.x;
  const int lane = tid & 31;
  const int wv   = tid >> 5;        // wave id 0..7 -> rows [wv*16, wv*16+16)
  const int mrow = lane & 15;       // A row / B column within fragment
  const int half = lane >> 4;

  const _Float16* fbase = fh + ((size_t)n1 * PC + (size_t)pt * BM) * DC;
  const _Float16* gbase = gh + (size_t)n2 * PC * DC;

  const unsigned int ldsA0 = (unsigned int)(uintptr_t)&As[0][0];
  const unsigned int ldsA1 = (unsigned int)(uintptr_t)&As[1][0];
  const unsigned int ldsB0 = (unsigned int)(uintptr_t)&Bs[0][0];
  const unsigned int ldsB1 = (unsigned int)(uintptr_t)&Bs[1][0];

  float rowmin[8];
  #pragma unroll
  for (int r = 0; r < 8; ++r) rowmin[r] = 1e30f;

  const v8f vzero = {};

  for (int qt = 0; qt < 8; ++qt) {
    v8f acc[8];
    #pragma unroll
    for (int j = 0; j < 8; ++j) acc[j] = vzero;

    const _Float16* gtile = gbase + (size_t)(qt * BN) * DC;

    // Prologue: DMA tile 0 into buffer 0.
    if (wv == 0) {
      tdm_load_tile32x128(fbase, ldsA0);
      tdm_load_tile32x128(gtile, ldsB0);
    }
    __builtin_amdgcn_s_wait_tensorcnt(0);
    __syncthreads();

    for (int ks = 0; ks < NK; ++ks) {
      const int cur = ks & 1;

      // Prefetch next tile into the other buffer while we compute.
      if (wv == 0 && ks + 1 < NK) {
        const int kk2 = (ks + 1) * BK;
        tdm_load_tile32x128(fbase + kk2, cur ? ldsA0 : ldsA1);
        tdm_load_tile32x128(gtile + kk2, cur ? ldsB0 : ldsB1);
      }

      // A fragment (16x32): lane m = lane&15,
      // K = [8*half..+7] ++ [16+8*half..+7]
      const _Float16* arow = &As[cur][(wv * 16 + mrow) * LDA];
      v8h alo = *(const v8h*)(arow + 8 * half);
      v8h ahi = *(const v8h*)(arow + 16 + 8 * half);
      v16h afrag = __builtin_shufflevector(alo, ahi,
                     0,1,2,3,4,5,6,7,8,9,10,11,12,13,14,15);

      #pragma unroll
      for (int j = 0; j < 8; ++j) {
        // B fragment (32x16): lane n = lane&15, K = [8*half..+7] ++ [16+8*half..+7]
        const _Float16* brow = &Bs[cur][(j * 16 + mrow) * LDA];
        v8h blo = *(const v8h*)(brow + 8 * half);
        v8h bhi = *(const v8h*)(brow + 16 + 8 * half);
        v16h bfrag = __builtin_shufflevector(blo, bhi,
                       0,1,2,3,4,5,6,7,8,9,10,11,12,13,14,15);
        acc[j] = __builtin_amdgcn_wmma_f32_16x16x32_f16(
                   false, afrag, false, bfrag, (short)0, acc[j], false, false);
      }

      __builtin_amdgcn_s_wait_tensorcnt(0);  // nonzero only on wave 0
      __syncthreads();
    }

    // Fold min over this q-tile. C layout: VGPR r holds row (r + 8*half),
    // column n = lane&15.
    const float* gb = gb2 + n2 * PC + qt * BN;
    float gbv[8];
    #pragma unroll
    for (int j = 0; j < 8; ++j) gbv[j] = gb[j * 16 + mrow];
    #pragma unroll
    for (int r = 0; r < 8; ++r) {
      float tmin = 1e30f;
      #pragma unroll
      for (int j = 0; j < 8; ++j)
        tmin = fminf(tmin, gbv[j] - 2.0f * acc[j][r]);
      #pragma unroll
      for (int off = 1; off < 16; off <<= 1)
        tmin = fminf(tmin, __shfl_xor(tmin, off, 32));
      rowmin[r] = fminf(rowmin[r], tmin);
    }
  }

  if (mrow == 0) {
    const float* fa  = fa2 + n1 * PC + pt * BM;
    float* outp = patch + (size_t)(n1 * N2C + n2) * PC + pt * BM;
    #pragma unroll
    for (int r = 0; r < 8; ++r) {
      int pl = wv * 16 + r + 8 * half;
      float d2 = fa[pl] + rowmin[r];
      outp[pl] = 0.5f * sqrtf(fmaxf(d2, 0.f));
    }
  }
}

// ---------------------------------------------------------------------------
// Kernel 3: per n1 -> scores[n1] = mean_m max_p patch[n1,m,p]
//           spm[n1,p] = mean_m patch[n1,m,p]
// ---------------------------------------------------------------------------
__global__ void reduce_kernel(const float* __restrict__ patch,
                              float* __restrict__ spm,
                              float* __restrict__ scores) {
  const int n1 = blockIdx.x;
  const int p  = threadIdx.x;          // 0..1023
  __shared__ float red[PC];
  float vals[N2C];
  float s = 0.f;
  #pragma unroll
  for (int m = 0; m < N2C; ++m) {
    vals[m] = patch[(size_t)(n1 * N2C + m) * PC + p];
    s += vals[m];
  }
  spm[n1 * PC + p] = s * (1.0f / N2C);

  float accm = 0.f;
  for (int m = 0; m < N2C; ++m) {
    red[p] = vals[m];
    __syncthreads();
    for (int off = PC / 2; off > 0; off >>= 1) {
      if (p < off) red[p] = fmaxf(red[p], red[p + off]);
      __syncthreads();
    }
    if (p == 0) accm += red[0];
    __syncthreads();
  }
  if (p == 0) scores[n1] = accm * (1.0f / N2C);
}

// ---------------------------------------------------------------------------
// Kernel 4: bilinear 32x32 -> 512x512 (half-pixel centers, edge clamp).
// ---------------------------------------------------------------------------
__global__ void bilinear_kernel(const float* __restrict__ spm,
                                float* __restrict__ outpix) {
  const int idx = blockIdx.x * blockDim.x + threadIdx.x;
  const int n   = idx >> 18;            // 512*512 = 2^18
  const int rem = idx & (HC * HC - 1);
  const int y   = rem >> 9;
  const int x   = rem & (HC - 1);
  const float scale = (float)PHC / (float)HC;  // 1/16
  float sy = ((float)y + 0.5f) * scale - 0.5f;
  float sx = ((float)x + 0.5f) * scale - 0.5f;
  float y0f = floorf(sy), x0f = floorf(sx);
  float wy = sy - y0f, wx = sx - x0f;
  int y0 = min(max((int)y0f, 0), PHC - 1);
  int y1 = min(max((int)y0f + 1, 0), PHC - 1);
  int x0 = min(max((int)x0f, 0), PHC - 1);
  int x1 = min(max((int)x0f + 1, 0), PHC - 1);
  const float* sp = spm + n * PC;
  float v00 = sp[y0 * PHC + x0];
  float v01 = sp[y0 * PHC + x1];
  float v10 = sp[y1 * PHC + x0];
  float v11 = sp[y1 * PHC + x1];
  float top = v00 * (1.f - wx) + v01 * wx;
  float bot = v10 * (1.f - wx) + v11 * wx;
  outpix[idx] = top * (1.f - wy) + bot * wy;
}

// ---------------------------------------------------------------------------
// Launch. Workspace ~25.5 MB: fh | gh | fa2 | gb2 | patch | spm
// ---------------------------------------------------------------------------
extern "C" void kernel_launch(void* const* d_in, const int* in_sizes, int n_in,
                              void* d_out, int out_size, void* d_ws, size_t ws_size,
                              hipStream_t stream) {
  const float* feats  = (const float*)d_in[0];
  const float* nfeats = (const float*)d_in[1];
  float* out = (float*)d_out;   // out[0..7] = scores, out[8..] = scores_pixel

  unsigned char* ws = (unsigned char*)d_ws;
  const size_t HBYTES = (size_t)N1C * PC * DC * sizeof(_Float16);
  _Float16* fh = (_Float16*)(ws);
  _Float16* gh = (_Float16*)(ws + HBYTES);
  float* fa2   = (float*)(ws + 2 * HBYTES);
  float* gb2   = fa2 + (size_t)N1C * PC;
  float* patch = gb2 + (size_t)N2C * PC;
  float* spm   = patch + (size_t)N1C * N2C * PC;

  hipLaunchKernelGGL(normalize_rows, dim3(N1C * PC), dim3(256), 0, stream,
                     feats, fh, fa2);
  hipLaunchKernelGGL(normalize_rows, dim3(N2C * PC), dim3(256), 0, stream,
                     nfeats, gh, gb2);
  hipLaunchKernelGGL(gemm_min_kernel, dim3(PC / BM, N1C * N2C), dim3(256), 0, stream,
                     fh, gh, fa2, gb2, patch);
  hipLaunchKernelGGL(reduce_kernel, dim3(N1C), dim3(PC), 0, stream,
                     patch, spm, out);
  hipLaunchKernelGGL(bilinear_kernel, dim3((N1C * HC * HC) / 256), dim3(256), 0, stream,
                     spm, out + 8);
}
